// DiffAttention_77489799954669
// MI455X (gfx1250) — matrix operand order repaired and checked by reading, not compiled, not run on hardware
//
#include <hip/hip_runtime.h>

// ---------------- problem dims ----------------
constexpr int B_ = 2, T_ = 1024, D_ = 2048, H_ = 16, KV_ = 4, Dh_ = 128;
constexpr int QCOLS = 2 * H_ * Dh_;   // 4096
constexpr int KCOLS = 2 * KV_ * Dh_;  // 1024
constexpr float SCALE = 0.08838834764831845f;  // 1/sqrt(128)
constexpr float LAMB  = 0.2f;                  // 0.8 - 0.6*exp(0)

// ---------------- types ----------------
typedef __attribute__((ext_vector_type(16))) _Float16 v16h;
typedef __attribute__((ext_vector_type(8)))  float    v8f;
typedef __attribute__((ext_vector_type(4)))  int      v4i32;

struct alignas(16) U4 { unsigned int x, y, z, w; };
union V16 { v16h v; U4 u[2]; };
union H8  { U4 u; _Float16 h[8]; };

// ---- async global->LDS (gfx1250 GLOBAL_LOAD_ASYNC_TO_LDS_B128), guarded ----
#if defined(__has_builtin)
#if __has_builtin(__builtin_amdgcn_global_load_async_to_lds_b128) && \
    __has_builtin(__builtin_amdgcn_s_wait_asynccnt)
#define USE_ASYNC_LDS 1
#endif
#endif
#ifndef USE_ASYNC_LDS
#define USE_ASYNC_LDS 0
#endif

__device__ __forceinline__ void cp16(const _Float16* g, _Float16* l) {
#if USE_ASYNC_LDS
    __builtin_amdgcn_global_load_async_to_lds_b128(
        (__attribute__((address_space(1))) v4i32*)g,
        (__attribute__((address_space(3))) v4i32*)l, 0, 0);
#else
    *(U4*)l = *(const U4*)g;
#endif
}

__device__ __forceinline__ void async_join() {
#if USE_ASYNC_LDS
    __builtin_amdgcn_s_wait_asynccnt(0);
#endif
}

__device__ __forceinline__ v8f vzero8() {
    v8f z;
    #pragma unroll
    for (int i = 0; i < 8; ++i) z[i] = 0.0f;
    return z;
}

// ---------------- fp32 -> f16 convert ----------------
__global__ void f32_to_f16_kernel(const float* __restrict__ in,
                                  _Float16* __restrict__ out, int n) {
    int i = blockIdx.x * 256 + threadIdx.x;
    if (i < n) out[i] = (_Float16)in[i];
}

// ---------------- RoPE (in place on f16 buffer) ----------------
__global__ void rope_kernel(_Float16* __restrict__ p, int nheads, int nrows) {
    int idx = blockIdx.x * 256 + threadIdx.x;
    int total = nrows * nheads * 128;
    if (idx >= total) return;
    int i   = idx & 63;
    int s   = (idx >> 6) & 1;
    int hh  = (idx >> 7) % nheads;
    int row = idx / (128 * nheads);
    int t   = row % T_;
    int stride = nheads * 256;
    size_t base = (size_t)row * stride + hh * 256 + s * 128 + i;
    float x1 = (float)p[base];
    float x2 = (float)p[base + 64];
    float inv = __powf(10000.0f, -(float)i * (1.0f / 64.0f));
    float ang = (float)t * inv;
    float sn, cs;
    __sincosf(ang, &sn, &cs);
    p[base]      = (_Float16)(x1 * cs - x2 * sn);
    p[base + 64] = (_Float16)(x1 * sn + x2 * cs);
}

// ---------------- tiled WMMA GEMM: C = A(MxK,f16) * B(KxN,f16) ----------------
// 256 threads = 8 waves; block tile 128x128; wave tile 32x64; K-step 32.
template <bool OUT_F32>
__global__ void gemm_wmma_kernel(const _Float16* __restrict__ A,
                                 const _Float16* __restrict__ Bm,
                                 void* __restrict__ Cout,
                                 int M, int N, int K, int nTilesN) {
    __shared__ _Float16 sA[128][40];    // [m][k]
    __shared__ _Float16 sBt[128][40];   // [n][k] (B transposed)

    const int tid  = threadIdx.x;
    const int lane = tid & 31, wv = tid >> 5;
    const int ln   = lane & 15, hs = lane >> 4;
    const int wm   = wv >> 1,   wn = wv & 1;
    const int m0   = (blockIdx.x / nTilesN) * 128;
    const int n0   = (blockIdx.x % nTilesN) * 128;

    v8f acc[2][4];
    #pragma unroll
    for (int a = 0; a < 2; ++a)
        #pragma unroll
        for (int b = 0; b < 4; ++b) acc[a][b] = vzero8();

    for (int k0 = 0; k0 < K; k0 += 32) {
        // stage A tile 128x32 (async global->LDS when available)
        #pragma unroll
        for (int t = 0; t < 2; ++t) {
            int c = tid + t * 256;
            int row = c >> 2;
            int dc  = (c & 3) * 8;
            cp16(&A[(size_t)(m0 + row) * K + k0 + dc], &sA[row][dc]);
        }
        // stage B tile 32x128 transposed into sBt[n][k]
        #pragma unroll
        for (int t = 0; t < 2; ++t) {
            int c  = tid + t * 256;
            int br = c >> 4;
            int dn = (c & 15) * 8;
            H8 vv; vv.u = *(const U4*)&Bm[(size_t)(k0 + br) * N + n0 + dn];
            #pragma unroll
            for (int j = 0; j < 8; ++j) sBt[dn + j][br] = vv.h[j];
        }
        // prefetch next K-step tiles (global_prefetch_b8)
        if (k0 + 32 < K) {
            __builtin_prefetch(&A[(size_t)(m0 + (tid >> 1)) * K + k0 + 32], 0, 3);
            __builtin_prefetch(
                &Bm[(size_t)(k0 + 32 + (tid & 31)) * N + n0 + (tid >> 5) * 16],
                0, 3);
        }
        async_join();
        __syncthreads();

        V16 af[2];
        #pragma unroll
        for (int fm = 0; fm < 2; ++fm) {
            int row = wm * 32 + fm * 16 + ln;
            af[fm].u[0] = *(const U4*)&sA[row][hs * 8];
            af[fm].u[1] = *(const U4*)&sA[row][16 + hs * 8];
        }
        #pragma unroll
        for (int fn = 0; fn < 4; ++fn) {
            int col = wn * 64 + fn * 16 + ln;
            V16 bf;
            bf.u[0] = *(const U4*)&sBt[col][hs * 8];
            bf.u[1] = *(const U4*)&sBt[col][16 + hs * 8];
            #pragma unroll
            for (int fm = 0; fm < 2; ++fm)
                acc[fm][fn] = __builtin_amdgcn_wmma_f32_16x16x32_f16(
                    false, af[fm].v, false, bf.v, (short)0, acc[fm][fn],
                    false, false);
        }
        __syncthreads();
    }

    #pragma unroll
    for (int fm = 0; fm < 2; ++fm)
        #pragma unroll
        for (int fn = 0; fn < 4; ++fn)
            #pragma unroll
            for (int r = 0; r < 8; ++r) {
                int row = m0 + wm * 32 + fm * 16 + r + 8 * hs;
                int col = n0 + wn * 64 + fn * 16 + ln;
                if (OUT_F32)
                    ((float*)Cout)[(size_t)row * N + col] = acc[fm][fn][r];
                else
                    ((_Float16*)Cout)[(size_t)row * N + col] =
                        (_Float16)acc[fm][fn][r];
            }
}

// ---------------- differential flash attention ----------------
// Block = (b, h, 16-query tile); 64 threads = 2 waves.
// Wave w owns softmax stream w and V columns [w*128, w*128+128) for BOTH
// streams' P·V accumulation. P tiles / alpha / l flow through LDS.
__global__ __launch_bounds__(64, 1)
void diff_attn_kernel(const _Float16* __restrict__ q,
                      const _Float16* __restrict__ k,
                      const _Float16* __restrict__ v,
                      _Float16* __restrict__ o) {
    __shared__ _Float16 sQ[16][264];      // [qrow][256 dims]
    __shared__ _Float16 sK[32][264];      // [key][256 dims]
    __shared__ _Float16 sVt[256][40];     // [vcol][key]
    __shared__ _Float16 sP[2][16][40];    // [stream][qrow][key]
    __shared__ float    sAl[2][16];       // per-tile rescale factors
    __shared__ float    sL[2][16];        // final softmax denominators

    const int tid  = threadIdx.x;
    const int lane = tid & 31, w = tid >> 5;
    const int ln   = lane & 15, hs = lane >> 4;

    const int bid   = blockIdx.x;
    const int qt    = bid & 63;
    const int h     = (bid >> 6) & 15;
    const int b     = bid >> 10;
    const int g     = h >> 2;             // kv head (REP = 4)
    const int qbase = qt * 16;

    // stage Q tile 16x256 into LDS once
    #pragma unroll
    for (int t = 0; t < 8; ++t) {
        int c   = tid + t * 64;
        int row = c >> 5;
        int dc  = (c & 31) * 8;
        cp16(&q[(size_t)(b * T_ + qbase + row) * QCOLS + h * 256 + dc],
             &sQ[row][dc]);
    }

    v8f acc[2][8];
    float m_[8], l_[8];
    #pragma unroll
    for (int s = 0; s < 2; ++s)
        #pragma unroll
        for (int n = 0; n < 8; ++n) acc[s][n] = vzero8();
    #pragma unroll
    for (int r = 0; r < 8; ++r) { m_[r] = -1e30f; l_[r] = 0.0f; }

    const int ntiles = (qbase + 47) >> 5;
    for (int kt = 0; kt < ntiles; ++kt) {
        // stage K tile (row major, async when available)
        #pragma unroll
        for (int t = 0; t < 16; ++t) {
            int c = tid + t * 64;
            int row = c >> 5;
            int dc  = (c & 31) * 8;
            cp16(&k[(size_t)(b * T_ + kt * 32 + row) * KCOLS + g * 256 + dc],
                 &sK[row][dc]);
        }
        // stage V tile transposed (manual scatter)
        #pragma unroll
        for (int t = 0; t < 16; ++t) {
            int c = tid + t * 64;
            int row = c >> 5;
            int dc  = (c & 31) * 8;
            H8 vv; vv.u = *(const U4*)
                &v[(size_t)(b * T_ + kt * 32 + row) * KCOLS + g * 256 + dc];
            #pragma unroll
            for (int j = 0; j < 8; ++j) sVt[dc + j][row] = vv.h[j];
        }
        // prefetch next tile (global_prefetch_b8)
        if (kt + 1 < ntiles) {
            size_t nrow = (size_t)(b * T_ + (kt + 1) * 32 + (tid >> 1)) * KCOLS
                          + g * 256 + (tid & 1) * 128;
            __builtin_prefetch(&k[nrow], 0, 3);
            __builtin_prefetch(&v[nrow], 0, 3);
        }
        async_join();
        __syncthreads();

        // scores + online softmax for OWNED stream w only
        {
            v8f s0 = vzero8(), s1 = vzero8();
            #pragma unroll
            for (int kd = 0; kd < 4; ++kd) {
                int qc = w * 128 + kd * 32;
                V16 qa;
                qa.u[0] = *(const U4*)&sQ[ln][qc + 16 * hs];
                qa.u[1] = *(const U4*)&sQ[ln][qc + 16 * hs + 8];
                int scol = w * 128 + kd * 32 + 16 * hs;
                V16 b0, b1;
                b0.u[0] = *(const U4*)&sK[ln][scol];
                b0.u[1] = *(const U4*)&sK[ln][scol + 8];
                b1.u[0] = *(const U4*)&sK[16 + ln][scol];
                b1.u[1] = *(const U4*)&sK[16 + ln][scol + 8];
                s0 = __builtin_amdgcn_wmma_f32_16x16x32_f16(
                    false, qa.v, false, b0.v, (short)0, s0, false, false);
                s1 = __builtin_amdgcn_wmma_f32_16x16x32_f16(
                    false, qa.v, false, b1.v, (short)0, s1, false, false);
            }
            #pragma unroll
            for (int r = 0; r < 8; ++r) {
                int qrow = qbase + r + 8 * hs;
                int c0 = kt * 32 + ln;
                float e0 = (c0      <= qrow) ? s0[r] * SCALE : -1e30f;
                float e1 = (c0 + 16 <= qrow) ? s1[r] * SCALE : -1e30f;
                float rm = fmaxf(e0, e1);
                #pragma unroll
                for (int off = 1; off < 16; off <<= 1)
                    rm = fmaxf(rm, __shfl_xor(rm, off, 32));
                float mn = fmaxf(m_[r], rm);
                float al = __expf(m_[r] - mn);
                float p0 = __expf(e0 - mn);
                float p1 = __expf(e1 - mn);
                float rs = p0 + p1;
                #pragma unroll
                for (int off = 1; off < 16; off <<= 1)
                    rs += __shfl_xor(rs, off, 32);
                l_[r] = l_[r] * al + rs;
                m_[r] = mn;
                sP[w][r + 8 * hs][ln]      = (_Float16)p0;
                sP[w][r + 8 * hs][16 + ln] = (_Float16)p1;
                if (ln == 0) sAl[w][r + 8 * hs] = al;
            }
        }
        __syncthreads();

        // rescale accumulators and P @ V for BOTH streams on this wave's V half
        #pragma unroll
        for (int s = 0; s < 2; ++s) {
            float al_r[8];
            #pragma unroll
            for (int r = 0; r < 8; ++r) al_r[r] = sAl[s][r + 8 * hs];
            #pragma unroll
            for (int n = 0; n < 8; ++n)
                #pragma unroll
                for (int r = 0; r < 8; ++r)
                    acc[s][n][r] = acc[s][n][r] * al_r[r];

            V16 pa;
            pa.u[0] = *(const U4*)&sP[s][ln][16 * hs];
            pa.u[1] = *(const U4*)&sP[s][ln][16 * hs + 8];
            #pragma unroll
            for (int n = 0; n < 8; ++n) {
                int vc = w * 128 + n * 16 + ln;
                V16 vb;
                vb.u[0] = *(const U4*)&sVt[vc][16 * hs];
                vb.u[1] = *(const U4*)&sVt[vc][16 * hs + 8];
                acc[s][n] = __builtin_amdgcn_wmma_f32_16x16x32_f16(
                    false, pa.v, false, vb.v, (short)0, acc[s][n], false, false);
            }
        }
        __syncthreads();
    }

    // publish per-stream denominators
    if (ln == 0) {
        #pragma unroll
        for (int r = 0; r < 8; ++r) sL[w][r + 8 * hs] = l_[r];
    }
    __syncthreads();

    // out = attn1@V / l1 - LAMB * attn2@V / l2
    #pragma unroll
    for (int n = 0; n < 8; ++n)
        #pragma unroll
        for (int r = 0; r < 8; ++r) {
            int row = qbase + r + 8 * hs;
            int col = h * 256 + w * 128 + n * 16 + ln;
            float l0 = sL[0][r + 8 * hs];
            float l1 = sL[1][r + 8 * hs];
            float ov = acc[0][n][r] / l0 - LAMB * acc[1][n][r] / l1;
            o[(size_t)(b * T_ + row) * QCOLS + col] = (_Float16)ov;
        }
}

// ---------------- launch ----------------
extern "C" void kernel_launch(void* const* d_in, const int* in_sizes, int n_in,
                              void* d_out, int out_size, void* d_ws,
                              size_t ws_size, hipStream_t stream) {
    (void)in_sizes; (void)n_in; (void)out_size; (void)ws_size;
    const float* x  = (const float*)d_in[0];
    const float* wq = (const float*)d_in[1];
    const float* wk = (const float*)d_in[2];
    const float* wv = (const float*)d_in[3];
    const float* wo = (const float*)d_in[4];

    const size_t MR = (size_t)B_ * T_;  // 2048 rows

    char* p = (char*)d_ws;
    auto alloc = [&](size_t halves) {
        _Float16* r = (_Float16*)p;
        p += halves * sizeof(_Float16);
        return r;
    };
    _Float16* xh  = alloc(MR * D_);
    _Float16* wqh = alloc((size_t)D_ * QCOLS);
    _Float16* wkh = alloc((size_t)D_ * KCOLS);
    _Float16* wvh = alloc((size_t)D_ * KCOLS);
    _Float16* woh = alloc((size_t)QCOLS * D_);
    _Float16* qh  = alloc(MR * QCOLS);
    _Float16* kh  = alloc(MR * KCOLS);
    _Float16* vh  = alloc(MR * KCOLS);
    _Float16* ah  = alloc(MR * QCOLS);

    auto cvt = [&](const float* in, _Float16* out, size_t n) {
        f32_to_f16_kernel<<<(unsigned)((n + 255) / 256), 256, 0, stream>>>(
            in, out, (int)n);
    };
    cvt(x,  xh,  MR * D_);
    cvt(wq, wqh, (size_t)D_ * QCOLS);
    cvt(wk, wkh, (size_t)D_ * KCOLS);
    cvt(wv, wvh, (size_t)D_ * KCOLS);
    cvt(wo, woh, (size_t)QCOLS * D_);

    // projections
    gemm_wmma_kernel<false><<<(MR / 128) * (QCOLS / 128), 256, 0, stream>>>(
        xh, wqh, qh, (int)MR, QCOLS, D_, QCOLS / 128);
    gemm_wmma_kernel<false><<<(MR / 128) * (KCOLS / 128), 256, 0, stream>>>(
        xh, wkh, kh, (int)MR, KCOLS, D_, KCOLS / 128);
    gemm_wmma_kernel<false><<<(MR / 128) * (KCOLS / 128), 256, 0, stream>>>(
        xh, wvh, vh, (int)MR, KCOLS, D_, KCOLS / 128);

    // RoPE on q1/q2 and k1/k2
    rope_kernel<<<(unsigned)((MR * H_ * 128 + 255) / 256), 256, 0, stream>>>(
        qh, H_, (int)MR);
    rope_kernel<<<(unsigned)((MR * KV_ * 128 + 255) / 256), 256, 0, stream>>>(
        kh, KV_, (int)MR);

    // differential attention
    diff_attn_kernel<<<B_ * H_ * (T_ / 16), 64, 0, stream>>>(qh, kh, vh, ah);

    // output projection (f32 out)
    gemm_wmma_kernel<true><<<(MR / 128) * (D_ / 128), 256, 0, stream>>>(
        ah, woh, d_out, (int)MR, D_, QCOLS, D_ / 128);
}
